// FeedForwardLoRA_30906584662346
// MI455X (gfx1250) — matrix operand order repaired and checked by reading, not compiled, not run on hardware
//
#include <hip/hip_runtime.h>

#define N_EMBD 1024
#define D_FF   4096
#define LORA_SCALE 0.25f   // alpha=1.0 / r=4

typedef __attribute__((ext_vector_type(16))) __bf16 v16bf;
typedef __attribute__((ext_vector_type(8)))  float  v8f;

// async global->LDS builtin expects pointers to a 16B int vector
typedef int v4i_async __attribute__((vector_size(16)));
typedef __attribute__((address_space(1))) v4i_async gas_v4i;   // global
typedef __attribute__((address_space(3))) v4i_async las_v4i;   // LDS

#if __has_builtin(__builtin_amdgcn_global_load_async_to_lds_b128)
#define USE_ASYNC_COPY 1
#else
#define USE_ASYNC_COPY 0
#endif

__device__ __forceinline__ void async_wait0() {
#if __has_builtin(__builtin_amdgcn_s_wait_asynccnt)
    __builtin_amdgcn_s_wait_asynccnt(0);
#else
    asm volatile("s_wait_asynccnt 0" ::: "memory");
#endif
}

__device__ __forceinline__ unsigned short f2bf(float f) {
    unsigned int u = __float_as_uint(f);
    unsigned int r = u + 0x7FFFu + ((u >> 16) & 1u);   // round-to-nearest-even
    return (unsigned short)(r >> 16);
}
__device__ __forceinline__ float bf2f(unsigned short b) {
    return __uint_as_float(((unsigned int)b) << 16);
}

// ---------------------------------------------------------------------------
// f32 -> bf16 conversion, 4 elements / thread
// ---------------------------------------------------------------------------
__global__ __launch_bounds__(256)
void cvt_f32_bf16(const float* __restrict__ src, unsigned short* __restrict__ dst,
                  size_t n) {
    size_t i = ((size_t)blockIdx.x * 256 + threadIdx.x) * 4;
    if (i + 3 < n) {
        float4 v = *(const float4*)(src + i);
        dst[i + 0] = f2bf(v.x);
        dst[i + 1] = f2bf(v.y);
        dst[i + 2] = f2bf(v.z);
        dst[i + 3] = f2bf(v.w);
    }
}

// ---------------------------------------------------------------------------
// rank-4 GEMV: T[m, 0..3] = sum_k S[m,k] * Aa[r,k]   (Aa = A_base[adapter])
// ---------------------------------------------------------------------------
__global__ __launch_bounds__(256)
void gemv_rank4_f32(const float* __restrict__ S, const float* __restrict__ A_base,
                    const int* __restrict__ adapter_id, float* __restrict__ T,
                    int M, int K) {
    int wave = threadIdx.x >> 5, lane = threadIdx.x & 31;
    int row = blockIdx.x * 8 + wave;
    if (row >= M) return;
    const float* Aa = A_base + (size_t)adapter_id[0] * 4 * K;
    const float* sr = S + (size_t)row * K;
    float s0 = 0.f, s1 = 0.f, s2 = 0.f, s3 = 0.f;
    for (int k = lane; k < K; k += 32) {
        float xv = sr[k];
        s0 += xv * Aa[k];
        s1 += xv * Aa[K + k];
        s2 += xv * Aa[2 * K + k];
        s3 += xv * Aa[3 * K + k];
    }
    for (int off = 16; off; off >>= 1) {
        s0 += __shfl_xor(s0, off, 32);
        s1 += __shfl_xor(s1, off, 32);
        s2 += __shfl_xor(s2, off, 32);
        s3 += __shfl_xor(s3, off, 32);
    }
    if (lane == 0) {
        float4 o = make_float4(s0, s1, s2, s3);
        *(float4*)(T + (size_t)row * 4) = o;
    }
}

__global__ __launch_bounds__(256)
void gemv_rank4_bf16(const unsigned short* __restrict__ S, const float* __restrict__ A_base,
                     const int* __restrict__ adapter_id, float* __restrict__ T,
                     int M, int K) {
    int wave = threadIdx.x >> 5, lane = threadIdx.x & 31;
    int row = blockIdx.x * 8 + wave;
    if (row >= M) return;
    const float* Aa = A_base + (size_t)adapter_id[0] * 4 * K;
    const unsigned short* sr = S + (size_t)row * K;
    float s0 = 0.f, s1 = 0.f, s2 = 0.f, s3 = 0.f;
    for (int k = lane; k < K; k += 32) {
        float xv = bf2f(sr[k]);
        s0 += xv * Aa[k];
        s1 += xv * Aa[K + k];
        s2 += xv * Aa[2 * K + k];
        s3 += xv * Aa[3 * K + k];
    }
    for (int off = 16; off; off >>= 1) {
        s0 += __shfl_xor(s0, off, 32);
        s1 += __shfl_xor(s1, off, 32);
        s2 += __shfl_xor(s2, off, 32);
        s3 += __shfl_xor(s3, off, 32);
    }
    if (lane == 0) {
        float4 o = make_float4(s0, s1, s2, s3);
        *(float4*)(T + (size_t)row * 4) = o;
    }
}

// ---------------------------------------------------------------------------
// Tiled WMMA GEMM with fused LoRA epilogue:
//   C[m,n] = sum_k A[m,k]*W[n,k]  + scale * dot(T[m,0..3], Bl[n,0..3])
// Block 256 thr (8 waves), tile 128x256, BK=32.
// Waves 2(m) x 4(n); each wave 64x64 = 4x4 WMMA tiles -> 16 wmma / K-step.
// Double-buffered LDS; global->LDS via async DMA (ASYNCcnt) when available.
// ---------------------------------------------------------------------------
#define BM 128
#define BN 256
#define BK 32
#define PITCH 40   // 32 + 8 bf16 pad (80B rows, 16B aligned)

__global__ __launch_bounds__(256)
void lora_gemm_bf16(const unsigned short* __restrict__ Abf,   // [M,K] bf16
                    const unsigned short* __restrict__ Wbf,   // [N,K] bf16
                    const float* __restrict__ Tlo,            // [M,4] f32
                    const float* __restrict__ Blo_base,       // [L,N,4] f32
                    const int* __restrict__ adapter_id,
                    unsigned short* __restrict__ out_bf16,    // nullable
                    float* __restrict__ out_f32,              // nullable
                    int M, int N, int K, int do_relu, float scale) {
    __shared__ unsigned short As[2][BM][PITCH];   // 20 KB
    __shared__ unsigned short Bs[2][BN][PITCH];   // 40 KB

    const int tid  = threadIdx.x;
    const int lane = tid & 31;
    const int wave = tid >> 5;
    const int wm   = wave & 1;   // m offset wm*64
    const int wn   = wave >> 1;  // n offset wn*64
    const int half = lane >> 4;  // 0/1
    const int r16  = lane & 15;

    const int m0 = blockIdx.y * BM;
    const int n0 = blockIdx.x * BN;

    // tile loaders: 16B chunks (8 ushorts). A: 512 chunks, B: 1024 chunks.
    int arow[2], acol[2], brow[4], bcol[4];
#pragma unroll
    for (int t = 0; t < 2; ++t) {
        int c = tid + t * 256;
        arow[t] = c >> 2; acol[t] = (c & 3) * 8;
    }
#pragma unroll
    for (int t = 0; t < 4; ++t) {
        int c = tid + t * 256;
        brow[t] = c >> 2; bcol[t] = (c & 3) * 8;
    }

    const int KT = K / BK;

    union Acc { v8f v; float f[8]; };
    Acc acc[4][4];
    v8f zero = {0.f, 0.f, 0.f, 0.f, 0.f, 0.f, 0.f, 0.f};
#pragma unroll
    for (int i = 0; i < 4; ++i)
#pragma unroll
        for (int j = 0; j < 4; ++j) acc[i][j].v = zero;

#if USE_ASYNC_COPY
    auto load_tile = [&](int kt, int buf) {
#pragma unroll
        for (int t = 0; t < 2; ++t)
            __builtin_amdgcn_global_load_async_to_lds_b128(
                (gas_v4i*)(Abf + (size_t)(m0 + arow[t]) * K + kt * BK + acol[t]),
                (las_v4i*)&As[buf][arow[t]][acol[t]],
                0, 0);
#pragma unroll
        for (int t = 0; t < 4; ++t)
            __builtin_amdgcn_global_load_async_to_lds_b128(
                (gas_v4i*)(Wbf + (size_t)(n0 + brow[t]) * K + kt * BK + bcol[t]),
                (las_v4i*)&Bs[buf][brow[t]][bcol[t]],
                0, 0);
    };
    load_tile(0, 0);
    async_wait0();
    __syncthreads();
#else
    uint4 pa[2], pb[4];
    auto gload = [&](int kt) {
#pragma unroll
        for (int t = 0; t < 2; ++t)
            pa[t] = *(const uint4*)(Abf + (size_t)(m0 + arow[t]) * K + kt * BK + acol[t]);
#pragma unroll
        for (int t = 0; t < 4; ++t)
            pb[t] = *(const uint4*)(Wbf + (size_t)(n0 + brow[t]) * K + kt * BK + bcol[t]);
    };
    auto sstore = [&](int buf) {
#pragma unroll
        for (int t = 0; t < 2; ++t)
            *(uint4*)&As[buf][arow[t]][acol[t]] = pa[t];
#pragma unroll
        for (int t = 0; t < 4; ++t)
            *(uint4*)&Bs[buf][brow[t]][bcol[t]] = pb[t];
    };
    gload(0);
    sstore(0);
    __syncthreads();
#endif

    int buf = 0;
    for (int kt = 0; kt < KT; ++kt) {
#if USE_ASYNC_COPY
        if (kt + 1 < KT) load_tile(kt + 1, buf ^ 1);
#else
        if (kt + 1 < KT) gload(kt + 1);
#endif

        // Fragment layouts (ISA):
        // A 16x32: row = lane%16; elems 0-7 at k=half*8, elems 8-15 at k=16+half*8
        // B 32x16: col n = lane%16; elems 0-7 at k=half*8, elems 8-15 at k=16+half*8
        union Frag { uint4 u[2]; v16bf v; };
        Frag afr[4], bfr[4];
#pragma unroll
        for (int i = 0; i < 4; ++i) {
            int r = wm * 64 + i * 16 + r16;
            afr[i].u[0] = *(const uint4*)&As[buf][r][half * 8];
            afr[i].u[1] = *(const uint4*)&As[buf][r][16 + half * 8];
        }
#pragma unroll
        for (int j = 0; j < 4; ++j) {
            int r = wn * 64 + j * 16 + r16;
            bfr[j].u[0] = *(const uint4*)&Bs[buf][r][half * 8];
            bfr[j].u[1] = *(const uint4*)&Bs[buf][r][16 + half * 8];
        }
#pragma unroll
        for (int i = 0; i < 4; ++i)
#pragma unroll
            for (int j = 0; j < 4; ++j)
                acc[i][j].v = __builtin_amdgcn_wmma_f32_16x16x32_bf16(
                    false, afr[i].v, false, bfr[j].v, (short)0, acc[i][j].v,
                    false, false);

#if USE_ASYNC_COPY
        async_wait0();
        __syncthreads();
#else
        if (kt + 1 < KT) sstore(buf ^ 1);
        __syncthreads();
#endif
        buf ^= 1;
    }

    // epilogue: fused rank-4 LoRA (+relu, + dual-format store)
    const float* Bl = Blo_base + (size_t)adapter_id[0] * (size_t)N * 4;

    float4 blv[4];
#pragma unroll
    for (int j = 0; j < 4; ++j) {
        int col = n0 + wn * 64 + j * 16 + r16;
        blv[j] = *(const float4*)(Bl + (size_t)col * 4);
    }
#pragma unroll
    for (int i = 0; i < 4; ++i) {
#pragma unroll
        for (int e = 0; e < 8; ++e) {
            int row = m0 + wm * 64 + i * 16 + e + 8 * half;
            float4 tv = *(const float4*)(Tlo + (size_t)row * 4);
#pragma unroll
            for (int j = 0; j < 4; ++j) {
                int col = n0 + wn * 64 + j * 16 + r16;
                float v = acc[i][j].f[e] +
                          scale * (tv.x * blv[j].x + tv.y * blv[j].y +
                                   tv.z * blv[j].z + tv.w * blv[j].w);
                if (do_relu) v = fmaxf(v, 0.f);
                size_t idx = (size_t)row * N + col;
                if (out_bf16) out_bf16[idx] = f2bf(v);
                if (out_f32)  out_f32[idx]  = v;
            }
        }
    }
}

// ---------------------------------------------------------------------------
// launch
// ---------------------------------------------------------------------------
extern "C" void kernel_launch(void* const* d_in, const int* in_sizes, int n_in,
                              void* d_out, int out_size, void* d_ws, size_t ws_size,
                              hipStream_t stream) {
    const float* x  = (const float*)d_in[0];
    const float* W1 = (const float*)d_in[1];
    const float* A1 = (const float*)d_in[2];
    const float* B1 = (const float*)d_in[3];
    const float* W2 = (const float*)d_in[4];
    const float* A2 = (const float*)d_in[5];
    const float* B2 = (const float*)d_in[6];
    const int* adapter = (const int*)d_in[7];

    const int M = in_sizes[0] / N_EMBD;   // B*T = 16384

    // carve workspace (256B aligned slices)
    size_t off = 0;
    char* base = (char*)d_ws;
    auto carve = [&](size_t bytes) -> void* {
        void* p = base + off;
        off += (bytes + 255) & ~(size_t)255;
        return p;
    };
    unsigned short* Xb  = (unsigned short*)carve((size_t)M * N_EMBD * 2);
    unsigned short* W1b = (unsigned short*)carve((size_t)D_FF * N_EMBD * 2);
    unsigned short* W2b = (unsigned short*)carve((size_t)N_EMBD * D_FF * 2);
    unsigned short* Hb  = (unsigned short*)carve((size_t)M * D_FF * 2);
    float* T1 = (float*)carve((size_t)M * 4 * sizeof(float));
    float* T2 = (float*)carve((size_t)M * 4 * sizeof(float));

    // 1) f32 -> bf16 casts
    size_t nx = (size_t)M * N_EMBD;
    cvt_f32_bf16<<<(unsigned)(nx / 1024), 256, 0, stream>>>(x, Xb, nx);
    size_t nw1 = (size_t)D_FF * N_EMBD;
    cvt_f32_bf16<<<(unsigned)(nw1 / 1024), 256, 0, stream>>>(W1, W1b, nw1);
    size_t nw2 = (size_t)N_EMBD * D_FF;
    cvt_f32_bf16<<<(unsigned)(nw2 / 1024), 256, 0, stream>>>(W2, W2b, nw2);

    // 2) t1 = x @ A1[ad]^T   [M,4]
    gemv_rank4_f32<<<M / 8, 256, 0, stream>>>(x, A1, adapter, T1, M, N_EMBD);

    // 3) h = relu(x @ W1^T + 0.25 * t1 @ B1[ad]^T)  -> bf16
    lora_gemm_bf16<<<dim3(D_FF / BN, M / BM), 256, 0, stream>>>(
        Xb, W1b, T1, B1, adapter, Hb, nullptr, M, D_FF, N_EMBD, 1, LORA_SCALE);

    // 4) t2 = h @ A2[ad]^T   [M,4]
    gemv_rank4_bf16<<<M / 8, 256, 0, stream>>>(Hb, A2, adapter, T2, M, D_FF);

    // 5) out = h @ W2^T + 0.25 * t2 @ B2[ad]^T  -> f32 d_out
    lora_gemm_bf16<<<dim3(N_EMBD / BN, M / BM), 256, 0, stream>>>(
        Hb, W2b, T2, B2, adapter, nullptr, (float*)d_out, M, N_EMBD, D_FF, 0, LORA_SCALE);
}
